// LinearAttention_14740327760100
// MI455X (gfx1250) — matrix operand order repaired and checked by reading, not compiled, not run on hardware
//
#include <hip/hip_runtime.h>

typedef __attribute__((ext_vector_type(16))) __bf16 v16bf;
typedef __attribute__((ext_vector_type(8)))  __bf16 v8bf;
typedef __attribute__((ext_vector_type(4)))  __bf16 v4bf;
typedef __attribute__((ext_vector_type(8)))  float  v8f;

__device__ __forceinline__ float fmap(float x) {
    // elu(x) + 1
    return x > 0.0f ? x + 1.0f : __expf(x);
}

__device__ __forceinline__ v16bf load_frag(const __bf16* p0, const __bf16* p1) {
    union { v16bf v; v8bf h[2]; } u;
    u.h[0] = *reinterpret_cast<const v8bf*>(p0);
    u.h[1] = *reinterpret_cast<const v8bf*>(p1);
    return u.v;
}

__device__ __forceinline__ v8f wmma_bf16(v16bf a, v16bf b, v8f c) {
    return __builtin_amdgcn_wmma_f32_16x16x32_bf16(
        /*neg_a=*/false, a, /*neg_b=*/false, b,
        /*c_mod=*/(short)0, c, /*reuse_a=*/false, /*reuse_b=*/false);
}

// Problem constants (from reference setup_inputs)
#define N_     4
#define L_     8192
#define S_     8192
#define H_     8
#define D_     64
#define NH     (N_ * H_)         // 32
#define PART_STRIDE 4160         // 64*64 KV + 64 ksum floats per partial

// ---------------------------------------------------------------------------
// Kernel 1: per (n,h,chunk) partial KV = phi(K)^T V  and  ksum = sum phi(K)
// ---------------------------------------------------------------------------
__global__ void __launch_bounds__(256)
kv_partial_kernel(const float* __restrict__ k, const float* __restrict__ v,
                  const float* __restrict__ kv_mask, float* __restrict__ part,
                  int nchunks, int chunkLen) {
    __shared__ __bf16 kT[64 * 40];   // [d][s] transposed, stride 40 (16B-aligned frags)
    __shared__ __bf16 vT[64 * 40];   // [e][s] transposed
    __shared__ float  red4[256 * 4];

    const int t     = threadIdx.x;
    const int lane  = t & 31;
    const int w     = t >> 5;        // wave 0..7
    const int chunk = blockIdx.x % nchunks;
    const int nh    = blockIdx.x / nchunks;
    const int n     = nh / H_;
    const int h     = nh % H_;

    // loader: thread owns d-quad [d4, d4+3], row groups rg and rg+16
    const int d4 = (t & 15) * 4;
    const int rg = t >> 4;           // 0..15

    const size_t baseNH = (size_t)n * S_ * H_ * D_ + (size_t)h * D_;
    const size_t mbase  = (size_t)n * S_;

    // wave -> two 16x16 tiles of the 64x64 output
    const int m      = lane & 15;
    const int half   = lane >> 4;
    const int dTile  = w >> 1;
    const int eTile0 = (2 * w) & 3;  // tiles (dTile, eTile0) and (dTile, eTile0+1)

    float ks[4] = {0.0f, 0.0f, 0.0f, 0.0f};
    v8f acc0 = {}; v8f acc1 = {};

    const int sBeg = chunk * chunkLen;
    for (int s0 = sBeg; s0 < sBeg + chunkLen; s0 += 32) {
        // cooperative load: 32x64 f32 tiles of k and v (b128 per thread),
        // feature-map, bf16, transpose into LDS
#pragma unroll
        for (int i = 0; i < 2; ++i) {
            const int srow = rg + 16 * i;
            const int s    = s0 + srow;
            const float msk = kv_mask[mbase + s];
            const size_t rbase = baseNH + (size_t)s * H_ * D_ + d4;
            const float4 k4 = *reinterpret_cast<const float4*>(&k[rbase]);
            const float4 v4 = *reinterpret_cast<const float4*>(&v[rbase]);
            const float f0 = msk * fmap(k4.x);
            const float f1 = msk * fmap(k4.y);
            const float f2 = msk * fmap(k4.z);
            const float f3 = msk * fmap(k4.w);
            ks[0] += f0; ks[1] += f1; ks[2] += f2; ks[3] += f3;
            kT[(d4 + 0) * 40 + srow] = (__bf16)f0;
            kT[(d4 + 1) * 40 + srow] = (__bf16)f1;
            kT[(d4 + 2) * 40 + srow] = (__bf16)f2;
            kT[(d4 + 3) * 40 + srow] = (__bf16)f3;
            vT[(d4 + 0) * 40 + srow] = (__bf16)(msk * v4.x);
            vT[(d4 + 1) * 40 + srow] = (__bf16)(msk * v4.y);
            vT[(d4 + 2) * 40 + srow] = (__bf16)(msk * v4.z);
            vT[(d4 + 3) * 40 + srow] = (__bf16)(msk * v4.w);
        }
        __syncthreads();

        // A = K^T tile: row d = dTile*16+m, K(s) cols {half*8..+7, half*8+16..+23}
        const int ar = (dTile * 16 + m) * 40;
        v16bf a = load_frag(&kT[ar + half * 8], &kT[ar + half * 8 + 16]);
        // B = V tile: col e = eTile*16+m, K(s) rows {half*8..+7, half*8+16..+23}
        const int br0 = (eTile0 * 16 + m) * 40 + half * 8;
        const int br1 = ((eTile0 + 1) * 16 + m) * 40 + half * 8;
        v16bf b0 = load_frag(&vT[br0], &vT[br0 + 16]);
        v16bf b1 = load_frag(&vT[br1], &vT[br1 + 16]);
        acc0 = wmma_bf16(a, b0, acc0);
        acc1 = wmma_bf16(a, b1, acc1);
        __syncthreads();
    }

    // write partial KV (f32, [d][e] row-major) and ksum
    float* pKV = part + (size_t)(nh * nchunks + chunk) * PART_STRIDE;
#pragma unroll
    for (int j = 0; j < 8; ++j) {
        const int dr = dTile * 16 + j + 8 * half;
        pKV[dr * 64 + eTile0 * 16 + m]       = acc0[j];
        pKV[dr * 64 + (eTile0 + 1) * 16 + m] = acc1[j];
    }
    red4[t * 4 + 0] = ks[0];
    red4[t * 4 + 1] = ks[1];
    red4[t * 4 + 2] = ks[2];
    red4[t * 4 + 3] = ks[3];
    __syncthreads();
    if (t < 64) {
        // column d = t was accumulated by threads t' = (d>>2) + 16*g, slot (d&3)
        const int q4 = t >> 2, j = t & 3;
        float s = 0.0f;
#pragma unroll
        for (int g = 0; g < 16; ++g) s += red4[(q4 + 16 * g) * 4 + j];
        pKV[4096 + t] = s;
    }
}

// ---------------------------------------------------------------------------
// Kernel 2: reduce chunk partials -> KV^T (bf16, [e][d]) + ksum (f32)
// ---------------------------------------------------------------------------
__global__ void __launch_bounds__(256)
kv_reduce_kernel(const float* __restrict__ part, float* __restrict__ ksumF,
                 __bf16* __restrict__ kvT, int nchunks) {
    const int nh = blockIdx.x;
    const int t  = threadIdx.x;
    const float* p = part + (size_t)nh * nchunks * PART_STRIDE;
#pragma unroll
    for (int i = 0; i < 16; ++i) {
        const int idx = t + 256 * i;             // 0..4095 over KV
        float s = 0.0f;
        for (int c = 0; c < nchunks; ++c) s += p[(size_t)c * PART_STRIDE + idx];
        const int dd = idx >> 6, ee = idx & 63;
        kvT[(size_t)nh * 4096 + ee * 64 + dd] = (__bf16)s;   // transposed store
    }
    if (t < 64) {
        float s = 0.0f;
        for (int c = 0; c < nchunks; ++c) s += p[(size_t)c * PART_STRIDE + 4096 + t];
        ksumF[nh * 64 + t] = s;
    }
}

// ---------------------------------------------------------------------------
// Kernel 3: out = (phi(Q) @ KV) / (phi(Q) . ksum + eps)
// ---------------------------------------------------------------------------
__global__ void __launch_bounds__(256)
attn_out_kernel(const float* __restrict__ q, const float* __restrict__ q_mask,
                const float* __restrict__ ksumF, const __bf16* __restrict__ kvT,
                float* __restrict__ out) {
    constexpr int ROWS = 128;
    __shared__ __bf16 qT[ROWS * 72];    // [row][d], stride 72
    __shared__ __bf16 kvL[64 * 72];     // [e][d], stride 72
    __shared__ float  ksumL[64];
    __shared__ float  divL[ROWS];

    const int t    = threadIdx.x;
    const int lane = t & 31;
    const int w    = t >> 5;
    const int rb   = blockIdx.x % (L_ / ROWS);
    const int nh   = blockIdx.x / (L_ / ROWS);
    const int n    = nh / H_;
    const int h    = nh % H_;

    // stage KV^T into LDS (u32 copies), and ksum
    {
        const uint32_t* src = reinterpret_cast<const uint32_t*>(kvT + (size_t)nh * 4096);
        uint32_t* dst = reinterpret_cast<uint32_t*>(kvL);
#pragma unroll
        for (int i = 0; i < 8; ++i) {
            const int idx = t + 256 * i;         // 0..2047 u32 words
            const int e = idx >> 5, dp = idx & 31;
            dst[e * 36 + dp] = src[idx];         // LDS row stride = 36 u32 = 72 bf16
        }
        if (t < 64) ksumL[t] = ksumF[nh * 64 + t];
    }

    // load q tile (128 x 64) with b128 loads, feature-map, bf16
    const int d4 = (t & 15) * 4;
    const int rg = t >> 4;               // 0..15
    const size_t baseNH = (size_t)n * L_ * H_ * D_ + (size_t)h * D_;
#pragma unroll
    for (int i = 0; i < 8; ++i) {
        const int row = rg + 16 * i;
        const int l   = rb * ROWS + row;
        const float msk = q_mask[(size_t)n * L_ + l];
        const float4 q4 = *reinterpret_cast<const float4*>(
            &q[baseNH + (size_t)l * H_ * D_ + d4]);
        v4bf pk;
        pk[0] = (__bf16)(msk * fmap(q4.x));
        pk[1] = (__bf16)(msk * fmap(q4.y));
        pk[2] = (__bf16)(msk * fmap(q4.z));
        pk[3] = (__bf16)(msk * fmap(q4.w));
        *reinterpret_cast<v4bf*>(&qT[row * 72 + d4]) = pk;   // 8B aligned
    }
    __syncthreads();

    // per-row normalizer
    if (t < ROWS) {
        float s = 0.0f;
#pragma unroll
        for (int dd = 0; dd < 64; ++dd) s += (float)qT[t * 72 + dd] * ksumL[dd];
        divL[t] = 1.0f / (s + 1e-6f);
    }
    __syncthreads();

    // wave w computes row-tile w (16 rows) x 4 e-tiles, K = 64 in two steps of 32
    const int m = lane & 15, half = lane >> 4;
    v8f acc[4] = {v8f{}, v8f{}, v8f{}, v8f{}};
#pragma unroll
    for (int ks = 0; ks < 2; ++ks) {
        const int ar = (w * 16 + m) * 72 + ks * 32 + half * 8;
        v16bf a = load_frag(&qT[ar], &qT[ar + 16]);
#pragma unroll
        for (int et = 0; et < 4; ++et) {
            const int br = (et * 16 + m) * 72 + ks * 32 + half * 8;
            v16bf b = load_frag(&kvL[br], &kvL[br + 16]);
            acc[et] = wmma_bf16(a, b, acc[et]);
        }
    }

    // scale by div and store f32 output [N,L,H,Dv]
#pragma unroll
    for (int j = 0; j < 8; ++j) {
        const int rloc = w * 16 + j + 8 * half;
        const int l    = rb * ROWS + rloc;
        const float dv = divL[rloc];
        const size_t ob = ((size_t)n * L_ + l) * H_ * 64 + (size_t)h * 64;
#pragma unroll
        for (int et = 0; et < 4; ++et)
            out[ob + et * 16 + m] = acc[et][j] * dv;
    }
}

// ---------------------------------------------------------------------------
extern "C" void kernel_launch(void* const* d_in, const int* in_sizes, int n_in,
                              void* d_out, int out_size, void* d_ws, size_t ws_size,
                              hipStream_t stream) {
    const float* q       = (const float*)d_in[0];
    const float* k       = (const float*)d_in[1];
    const float* v       = (const float*)d_in[2];
    const float* q_mask  = (const float*)d_in[3];
    const float* kv_mask = (const float*)d_in[4];
    float* out = (float*)d_out;

    // choose the largest power-of-two chunk count (parallelism for kernel 1)
    // that fits the workspace; deterministic given ws_size.
    int nchunks = 32;
    while (nchunks > 1) {
        size_t need = (size_t)NH * nchunks * PART_STRIDE * sizeof(float)
                    + (size_t)NH * 64 * sizeof(float)
                    + (size_t)NH * 4096 * 2;
        if (need <= ws_size) break;
        nchunks >>= 1;
    }
    const int chunkLen = S_ / nchunks;

    float*  part  = (float*)d_ws;
    float*  ksumF = part + (size_t)NH * nchunks * PART_STRIDE;
    __bf16* kvT   = reinterpret_cast<__bf16*>(ksumF + NH * 64);

    kv_partial_kernel<<<NH * nchunks, 256, 0, stream>>>(k, v, kv_mask, part,
                                                        nchunks, chunkLen);
    kv_reduce_kernel<<<NH, 256, 0, stream>>>(part, ksumF, kvT, nchunks);
    attn_out_kernel<<<NH * (L_ / 128), 256, 0, stream>>>(q, q_mask, ksumF, kvT, out);
}
